// GINLinkPred_4337916969093
// MI455X (gfx1250) — matrix operand order repaired and checked by reading, not compiled, not run on hardware
//
#include <hip/hip_runtime.h>
#include <hip/hip_bf16.h>

#define N_NODES 100000
#define N_EDGES 1600000
#define N_POS   100000
#define IN_C 128
#define HID  256
#define OUT_C 128

typedef __attribute__((ext_vector_type(16))) __bf16 v16bf;
typedef __attribute__((ext_vector_type(8)))  float  v8f;

__device__ __forceinline__ unsigned short f2bf_bits(float f) {
  union { float f; unsigned u; } c; c.f = f;
  // round-to-nearest-even f32 -> bf16
  return (unsigned short)((c.u + 0x7FFFu + ((c.u >> 16) & 1u)) >> 16);
}

union AFrag { v16bf v; unsigned short u[16]; unsigned u32[8]; };
union BFrag { v16bf v; unsigned short u[16]; uint4 q[2]; };

// ---------------------------------------------------------------------------
// 1) h = x  (agg accumulator seeded with x, since GIN eps=0 => h = x + sum)
// ---------------------------------------------------------------------------
__global__ void init_h_kernel(const float4* __restrict__ x4, float4* __restrict__ h4) {
  int idx = blockIdx.x * blockDim.x + threadIdx.x;
  if (idx < N_NODES * (IN_C / 4)) h4[idx] = x4[idx];
}

// ---------------------------------------------------------------------------
// 2) Repack fp32 row-major weights [K][N] into bf16 WMMA B-fragment-major
//    layout: dst[((kt*(N/16)+nt)*32 + lane)*16 + e] = bf16(W[k][n])
//    with k = kt*32 + (e<8 ? (lane>>4)*8 + e : 16 + (lane>>4)*8 + (e-8)), n = nt*16 + (lane&15)
//    (per CDNA5 ISA 32x16 16-bit B layout; lanes 0-15 K=0..7,16..23, 16-31 K=8..15,24..31)
// ---------------------------------------------------------------------------
__global__ void wconv_kernel(const float* __restrict__ src, unsigned short* __restrict__ dst,
                             int K, int N) {
  int idx = blockIdx.x * blockDim.x + threadIdx.x;
  int total = (K >> 5) * (N >> 4) * 512;
  if (idx >= total) return;
  int e = idx & 15;
  int l = (idx >> 4) & 31;
  int f = idx >> 9;
  int ntiles = N >> 4;
  int nt = f % ntiles, kt = f / ntiles;
  int k = kt * 32 + (e < 8 ? (l >> 4) * 8 + e : 16 + (l >> 4) * 8 + (e - 8));
  int n = nt * 16 + (l & 15);
  dst[idx] = f2bf_bits(src[(size_t)k * N + n]);
}

// ---------------------------------------------------------------------------
// 3) Edge scatter-add: h[dst] += x[src].  One wave per edge, float4 per lane.
//    x and h are L2-resident (51.2 MB each vs 192 MB L2) -> L2-atomic bound.
// ---------------------------------------------------------------------------
__global__ void edge_agg_kernel(const float4* __restrict__ x4,
                                const int* __restrict__ ei,
                                float* __restrict__ h) {
  long long idx = (long long)blockIdx.x * blockDim.x + threadIdx.x;
  if (idx >= (long long)N_EDGES * 32) return;
  int e = (int)(idx >> 5);
  int g = (int)(idx & 31);
  int src = ei[e];
  int dst = ei[N_EDGES + e];
  float4 v = x4[(size_t)src * (IN_C / 4) + g];
  float* p = h + (size_t)dst * IN_C + (g << 2);
  unsafeAtomicAdd(p + 0, v.x);   // -> global_atomic_add_f32
  unsafeAtomicAdd(p + 1, v.y);
  unsafeAtomicAdd(p + 2, v.z);
  unsafeAtomicAdd(p + 3, v.w);
}

// ---------------------------------------------------------------------------
// 4) One-pass h(f32) -> hbf(bf16) row-major conversion, so the GEMM1 inner
//    loop stays pure global_load + v_wmma (no per-element cvt VALU chains,
//    no 4x-redundant conversion across waves sharing an M-tile).
// ---------------------------------------------------------------------------
__global__ void hconv_kernel(const float4* __restrict__ h4, ushort4* __restrict__ hbf4) {
  int idx = blockIdx.x * blockDim.x + threadIdx.x;
  if (idx >= N_NODES * (IN_C / 4)) return;
  float4 v = h4[idx];
  ushort4 o;
  o.x = f2bf_bits(v.x);
  o.y = f2bf_bits(v.y);
  o.z = f2bf_bits(v.z);
  o.w = f2bf_bits(v.w);
  hbf4[idx] = o;
}

// ---------------------------------------------------------------------------
// 5) hid = relu(hbf @ W1 + b1), M=100000 K=128 N=256, bf16 WMMA, f32 accum.
//    Block = 256 thr = 8 waves, covers 32 rows x 256 cols.
//    Wave w: mtile = w>>2 (16 rows), n-quad = w&3 (4 ntiles = 64 cols).
// ---------------------------------------------------------------------------
__global__ __launch_bounds__(256) void gemm1_relu_kernel(
    const unsigned short* __restrict__ hbf, const unsigned short* __restrict__ W1f,
    const float* __restrict__ b1, unsigned short* __restrict__ hid) {
  const int lane  = threadIdx.x & 31;
  const int w     = threadIdx.x >> 5;
  const int lmod  = lane & 15, lhalf = lane >> 4;
  const int mbase = blockIdx.x * 32 + (w >> 2) * 16;
  const int nq    = (w & 3) * 4;
  const int m     = mbase + lmod;
  const int koff  = lhalf * 8;

  v8f acc[4] = {};
  for (int kt = 0; kt < 4; ++kt) {
    // A fragment: 16x32 bf16 per ISA layout, packed-pair loads
    AFrag a;
    const unsigned short* arow = hbf + (size_t)m * IN_C + kt * 32;
#pragma unroll
    for (int p = 0; p < 8; ++p) {
      int K = (p < 4) ? (koff + 2 * p) : (16 + koff + 2 * (p - 4));
      a.u32[p] = *(const unsigned*)(arow + K);
    }
#pragma unroll
    for (int t = 0; t < 4; ++t) {
      int nt = nq + t;
      BFrag b;
      const uint4* src = (const uint4*)(W1f + ((size_t)((kt * (HID / 16) + nt) * 32 + lane) << 4));
      b.q[0] = src[0];
      b.q[1] = src[1];
      acc[t] = __builtin_amdgcn_wmma_f32_16x16x32_bf16(
          false, a.v, false, b.v, (short)0, acc[t], false, false);
    }
  }
  // epilogue: bias + relu, store bf16 (C/D layout: vgpr r -> M = r + lhalf*8)
  const int mrow = mbase + lhalf * 8;
#pragma unroll
  for (int t = 0; t < 4; ++t) {
    int n = (nq + t) * 16 + lmod;
    float bias = b1[n];
#pragma unroll
    for (int r = 0; r < 8; ++r) {
      float v = acc[t][r] + bias;
      hid[(size_t)(mrow + r) * HID + n] = f2bf_bits(v > 0.f ? v : 0.f);
    }
  }
}

// ---------------------------------------------------------------------------
// 6) z = hid @ W2 + b2, M=100000 K=256 N=128, bf16 WMMA, f32 accum/out.
//    Wave w: mtile = w>>2, n-pair = w&3 (2 ntiles = 32 cols).
// ---------------------------------------------------------------------------
__global__ __launch_bounds__(256) void gemm2_bias_kernel(
    const unsigned short* __restrict__ hid, const unsigned short* __restrict__ W2f,
    const float* __restrict__ b2, float* __restrict__ z) {
  const int lane  = threadIdx.x & 31;
  const int w     = threadIdx.x >> 5;
  const int lmod  = lane & 15, lhalf = lane >> 4;
  const int mbase = blockIdx.x * 32 + (w >> 2) * 16;
  const int nq    = (w & 3) * 2;
  const int m     = mbase + lmod;
  const int koff  = lhalf * 8;

  v8f acc[2] = {};
  for (int kt = 0; kt < 8; ++kt) {
    AFrag a;
    const unsigned short* arow = hid + (size_t)m * HID + kt * 32;
#pragma unroll
    for (int p = 0; p < 8; ++p) {
      int K = (p < 4) ? (koff + 2 * p) : (16 + koff + 2 * (p - 4));
      a.u32[p] = *(const unsigned*)(arow + K);  // two packed bf16
    }
#pragma unroll
    for (int t = 0; t < 2; ++t) {
      int nt = nq + t;
      BFrag b;
      const uint4* src = (const uint4*)(W2f + ((size_t)((kt * (OUT_C / 16) + nt) * 32 + lane) << 4));
      b.q[0] = src[0];
      b.q[1] = src[1];
      acc[t] = __builtin_amdgcn_wmma_f32_16x16x32_bf16(
          false, a.v, false, b.v, (short)0, acc[t], false, false);
    }
  }
  const int mrow = mbase + lhalf * 8;
#pragma unroll
  for (int t = 0; t < 2; ++t) {
    int n = (nq + t) * 16 + lmod;
    float bias = b2[n];
#pragma unroll
    for (int r = 0; r < 8; ++r)
      z[(size_t)(mrow + r) * OUT_C + n] = acc[t][r] + bias;
  }
}

// ---------------------------------------------------------------------------
// 7) logits[p] = dot(z[a], z[b]); one wave32 per pair, shfl_xor reduction.
// ---------------------------------------------------------------------------
__global__ void decode_kernel(const float* __restrict__ z, const int* __restrict__ pei,
                              float* __restrict__ out) {
  int gid = blockIdx.x * blockDim.x + threadIdx.x;
  int p = gid >> 5, lane = gid & 31;
  if (p >= N_POS) return;
  int a = pei[p], b = pei[N_POS + p];
  float4 za = *(const float4*)(z + (size_t)a * OUT_C + (lane << 2));
  float4 zb = *(const float4*)(z + (size_t)b * OUT_C + (lane << 2));
  float s = za.x * zb.x + za.y * zb.y + za.z * zb.z + za.w * zb.w;
#pragma unroll
  for (int off = 16; off > 0; off >>= 1) s += __shfl_xor(s, off, 32);
  if (lane == 0) out[p] = s;
}

// ---------------------------------------------------------------------------
// Host launch. Workspace layout (all 16B-aligned):
//   h   f32  [100000][128]   51,200,000 B @ 0          (reused as z after GEMM1)
//   hbf bf16 [100000][128]   25,600,000 B @ 51,200,000
//   hid bf16 [100000][256]   51,200,000 B @ 76,800,000
//   W1f bf16 fragment-major      65,536 B @ 128,000,000
//   W2f bf16 fragment-major      65,536 B @ 128,065,536
//   total ~128.1 MB.  z aliases h: h is dead after hconv, and gemm2 fully
//   rewrites z before decode reads it, so every replay is identical.
// ---------------------------------------------------------------------------
extern "C" void kernel_launch(void* const* d_in, const int* in_sizes, int n_in,
                              void* d_out, int out_size, void* d_ws, size_t ws_size,
                              hipStream_t stream) {
  const float* x  = (const float*)d_in[0];
  const int*   ei = (const int*)d_in[1];
  const int*   pei= (const int*)d_in[2];
  const float* W1 = (const float*)d_in[3];
  const float* b1 = (const float*)d_in[4];
  const float* W2 = (const float*)d_in[5];
  const float* b2 = (const float*)d_in[6];
  float* out = (float*)d_out;

  char* ws = (char*)d_ws;
  float*          h   = (float*)(ws);
  float*          z   = (float*)(ws);                        // alias of h
  unsigned short* hbf = (unsigned short*)(ws + 51200000ull);
  unsigned short* hid = (unsigned short*)(ws + 76800000ull);
  unsigned short* W1f = (unsigned short*)(ws + 128000000ull);
  unsigned short* W2f = (unsigned short*)(ws + 128065536ull);

  init_h_kernel<<<12500, 256, 0, stream>>>((const float4*)x, (float4*)h);
  wconv_kernel<<<128, 256, 0, stream>>>(W1, W1f, IN_C, HID);
  wconv_kernel<<<128, 256, 0, stream>>>(W2, W2f, HID, OUT_C);
  edge_agg_kernel<<<200000, 256, 0, stream>>>((const float4*)x, ei, h);
  hconv_kernel<<<12500, 256, 0, stream>>>((const float4*)h, (ushort4*)hbf);
  gemm1_relu_kernel<<<3125, 256, 0, stream>>>(hbf, W1f, b1, hid);
  gemm2_bias_kernel<<<3125, 256, 0, stream>>>(hid, W2f, b2, z);
  decode_kernel<<<12500, 256, 0, stream>>>(z, pei, out);
}
